// CLS_SSG_Model_79104707658392
// MI455X (gfx1250) — matrix operand order repaired and checked by reading, not compiled, not run on hardware
//
#include <hip/hip_runtime.h>

#define BATCH 8
#define NPTS  2048
#define ALPHA 0.2f

typedef __attribute__((ext_vector_type(16))) __bf16 bf16x16;
typedef __attribute__((ext_vector_type(8)))  float  f32x8;

static inline int cdiv(int a, int b) { return (a + b - 1) / b; }

// ---- CDNA5 async global->LDS copy (VGLOBAL GLOBAL_LOAD_ASYNC_TO_LDS_B32) ----
// Per-lane: LDS[lds_off] = MEM[gptr]; tracked by ASYNCcnt.
__device__ __forceinline__ void async_load_f32_to_lds(const float* gptr, float* lptr)
{
    unsigned lds_off = (unsigned)(reinterpret_cast<uintptr_t>(lptr));   // low 32 bits = LDS addr
    asm volatile("global_load_async_to_lds_b32 %0, %1, off"
                 :: "v"(lds_off), "v"(gptr) : "memory");
}
__device__ __forceinline__ void wait_async_lds()
{
    asm volatile("s_wait_asynccnt 0x0" ::: "memory");
}

// ---------------- Farthest point sampling (one block per batch) ----------------
__global__ void fps_kernel(const float* __restrict__ pts, int N, int npoint,
                           int* __restrict__ fidx, float* __restrict__ new_xyz)
{
    __shared__ float dist[2048];
    __shared__ float rv[256];
    __shared__ int   ri[256];
    const int b   = blockIdx.x;
    const int tid = threadIdx.x;
    const float* P = pts + (size_t)b * N * 3;
    for (int i = tid; i < N; i += 256) dist[i] = 1e10f;
    __syncthreads();
    int far = 0;
    for (int it = 0; it < npoint; ++it) {
        float fx = P[far * 3 + 0], fy = P[far * 3 + 1], fz = P[far * 3 + 2];
        if (tid == 0) {
            fidx[b * npoint + it] = far;
            float* o = new_xyz + ((size_t)b * npoint + it) * 3;
            o[0] = fx; o[1] = fy; o[2] = fz;
        }
        float bv = -1.0f; int bi = 0;
        for (int i = tid; i < N; i += 256) {
            float dx = P[i * 3 + 0] - fx;
            float dy = P[i * 3 + 1] - fy;
            float dz = P[i * 3 + 2] - fz;
            float d  = dx * dx + dy * dy + dz * dz;
            float nd = fminf(dist[i], d);
            dist[i] = nd;
            if (nd > bv) { bv = nd; bi = i; }   // strict > keeps lowest index (argmax-first)
        }
        rv[tid] = bv; ri[tid] = bi;
        __syncthreads();
        for (int s = 128; s > 0; s >>= 1) {
            if (tid < s) {
                if (rv[tid + s] > rv[tid] ||
                    (rv[tid + s] == rv[tid] && ri[tid + s] < ri[tid])) {
                    rv[tid] = rv[tid + s]; ri[tid] = ri[tid + s];
                }
            }
            __syncthreads();
        }
        far = ri[0];
        __syncthreads();
    }
}

// ---------------- Ball query: first nsample hits in index order ----------------
__global__ void ballq_kernel(const float* __restrict__ ctr, const float* __restrict__ pts,
                             int S, int N, float r2, int* __restrict__ idx)
{
    int g = blockIdx.x * blockDim.x + threadIdx.x;
    if (g >= BATCH * S) return;
    int b = g / S, s = g % S;
    const float* q = ctr + ((size_t)b * S + s) * 3;
    float qx = q[0], qy = q[1], qz = q[2];
    const float* P = pts + (size_t)b * N * 3;
    int* o = idx + ((size_t)b * S + s) * 32;
    int cnt = 0;
    for (int j = 0; j < N && cnt < 32; ++j) {
        float dx = qx - P[j * 3 + 0];
        float dy = qy - P[j * 3 + 1];
        float dz = qz - P[j * 3 + 2];
        if (dx * dx + dy * dy + dz * dz < r2) o[cnt++] = j;
    }
    int pad = (cnt > 0) ? o[0] : 0;
    for (; cnt < 32; ++cnt) o[cnt] = pad;
}

// ------- Gather + recenter into K-padded rows: [pts[idx]-ctr, feats[idx], 0..] --
__global__ void group_kernel(const float* __restrict__ pts, const float* __restrict__ feats,
                             const float* __restrict__ ctr, const int* __restrict__ idx,
                             int S, int N, int Cprev, int Cpad, float* __restrict__ out)
{
    int g = blockIdx.x * blockDim.x + threadIdx.x;
    if (g >= BATCH * S * 32) return;
    int b = g / (S * 32), r = g % (S * 32), s = r / 32;
    int j = idx[(size_t)g];
    int Cin = 3 + Cprev;
    float* o = out + (size_t)g * Cpad;
    const float* pj = pts + ((size_t)b * N + j) * 3;
    const float* c  = ctr + ((size_t)b * S + s) * 3;
    o[0] = pj[0] - c[0];
    o[1] = pj[1] - c[1];
    o[2] = pj[2] - c[2];
    if (feats) {
        const float* f = feats + ((size_t)b * N + j) * Cprev;
        for (int cc = 0; cc < Cprev; ++cc) o[3 + cc] = f[cc];
    }
    for (int cc = Cin; cc < Cpad; ++cc) o[cc] = 0.0f;   // zero the K padding
}

// ---------------- GEMM + bias + leaky-ReLU via v_wmma_f32_16x16x32_bf16 --------
// X[M,Cpad] * W[Ctrue,D] -> Y[M,D].  Launcher guarantees:
//   M % 64 == 0, Cpad % 32 == 0, D % 32 == 0, X columns >= Ctrue are zero.
// Each wave computes a 16x32 output tile (one A fragment, two B fragments, two
// accumulators); 4 waves stack rows -> 64x32 per block.  B tile (32x32) staged
// into LDS with GLOBAL_LOAD_ASYNC_TO_LDS_B32 (ASYNCcnt), bank-padded stride 33.
// W row index is clamped branch-free: padded X columns are zero, nullifying any
// clamped row's contribution.
__global__ void __launch_bounds__(128)
gemm_bias_lrelu(const float* __restrict__ X, const float* __restrict__ W,
                const float* __restrict__ Bv, float* __restrict__ Y,
                int Ctrue, int Cpad, int D)
{
    __shared__ float Wt[32 * 33];               // [k 0..31][col 0..31], stride 33

    const int tid  = threadIdx.x;
    const int lane = tid & 31;
    const int wave = tid >> 5;
    const int half = lane >> 4;
    const int l15  = lane & 15;
    const int m0 = blockIdx.x * 64 + wave * 16;
    const int n1 = blockIdx.y * 32;

    f32x8 acc0 = {}, acc1 = {};
    const float* xrow = X + (size_t)(m0 + l15) * Cpad;

    for (int kb = 0; kb < Cpad; kb += 32) {
        // ---- async-DMA B tile (32x32 f32) into LDS, coalesced ----
#pragma unroll
        for (int q = 0; q < 8; ++q) {
            int ii = tid + (q << 7);            // 0..1023
            int r  = ii >> 5, cc = ii & 31;
            int kk = min(kb + r, Ctrue - 1);    // branch-free clamp
            async_load_f32_to_lds(&W[(size_t)kk * D + n1 + cc], &Wt[r * 33 + cc]);
        }
        wait_async_lds();
        __syncthreads();

        // ---- A fragment: k runs [kb+8h, kb+8h+8) and [kb+16+8h, kb+16+8h+8) ----
        const float* abase = xrow + kb + (half << 3);
        float4 a0 = *(const float4*)(abase + 0);
        float4 a1 = *(const float4*)(abase + 4);
        float4 a2 = *(const float4*)(abase + 16);
        float4 a3 = *(const float4*)(abase + 20);
        bf16x16 a;
        a[0]  = (__bf16)a0.x; a[1]  = (__bf16)a0.y; a[2]  = (__bf16)a0.z; a[3]  = (__bf16)a0.w;
        a[4]  = (__bf16)a1.x; a[5]  = (__bf16)a1.y; a[6]  = (__bf16)a1.z; a[7]  = (__bf16)a1.w;
        a[8]  = (__bf16)a2.x; a[9]  = (__bf16)a2.y; a[10] = (__bf16)a2.z; a[11] = (__bf16)a2.w;
        a[12] = (__bf16)a3.x; a[13] = (__bf16)a3.y; a[14] = (__bf16)a3.z; a[15] = (__bf16)a3.w;

        // ---- two B fragments from LDS: k runs [8h, 8h+8) and [16+8h, 16+8h+8) ----
        bf16x16 b0, b1;
#pragma unroll
        for (int e = 0; e < 16; ++e) {
            int rr = (half << 3) + e + ((e >> 3) << 3);
            b0[e] = (__bf16)Wt[rr * 33 + l15];
            b1[e] = (__bf16)Wt[rr * 33 + 16 + l15];
        }

        acc0 = __builtin_amdgcn_wmma_f32_16x16x32_bf16(
                   false, a, false, b0, (short)0, acc0, false, false);
        acc1 = __builtin_amdgcn_wmma_f32_16x16x32_bf16(
                   false, a, false, b1, (short)0, acc1, false, false);
        __syncthreads();
    }

    float bias0 = Bv[n1 + l15];
    float bias1 = Bv[n1 + 16 + l15];
#pragma unroll
    for (int v = 0; v < 8; ++v) {
        int r = m0 + (half << 3) + v;           // C/D layout: row = v + 8*half
        float v0 = acc0[v] + bias0;
        float v1 = acc1[v] + bias1;
        v0 = (v0 >= 0.0f) ? v0 : ALPHA * v0;
        v1 = (v1 >= 0.0f) ? v1 : ALPHA * v1;
        Y[(size_t)r * D + n1 + l15]      = v0;
        Y[(size_t)r * D + n1 + 16 + l15] = v1;
    }
}

// ---------------- Max-pool over neighborhood K (float4 channels) ---------------
__global__ void maxpool_kernel(const float* __restrict__ x, int S, int K, int D,
                               float* __restrict__ y)
{
    int d4 = D >> 2;
    int g = blockIdx.x * blockDim.x + threadIdx.x;
    if (g >= BATCH * S * d4) return;
    int b = g / (S * d4), r = g % (S * d4), s = r / d4, d = (r % d4) << 2;
    const float* p = x + (((size_t)b * S + s) * K) * D + d;
    float4 m = *(const float4*)p;
    for (int k = 1; k < K; ++k) {
        float4 v = *(const float4*)(p + (size_t)k * D);
        m.x = fmaxf(m.x, v.x); m.y = fmaxf(m.y, v.y);
        m.z = fmaxf(m.z, v.z); m.w = fmaxf(m.w, v.w);
    }
    *(float4*)(y + ((size_t)b * S + s) * D + d) = m;
}

// ---------------- FP: 3-NN inverse-distance interpolation + skip concat --------
__global__ void fp_interp_kernel(const float* __restrict__ xyz1, const float* __restrict__ xyz2,
                                 const float* __restrict__ p2, const float* __restrict__ p1,
                                 int N1, int N2, int C2, int C1, float* __restrict__ out)
{
    int g = blockIdx.x * blockDim.x + threadIdx.x;
    if (g >= BATCH * N1) return;
    int b = g / N1, n = g % N1;
    const float* q = xyz1 + ((size_t)b * N1 + n) * 3;
    float qx = q[0], qy = q[1], qz = q[2];
    const float* P = xyz2 + (size_t)b * N2 * 3;
    float bd0 = 1e30f, bd1 = 1e30f, bd2 = 1e30f;
    int   bi0 = 0,     bi1 = 0,     bi2 = 0;
    for (int j = 0; j < N2; ++j) {
        float dx = qx - P[j * 3 + 0];
        float dy = qy - P[j * 3 + 1];
        float dz = qz - P[j * 3 + 2];
        float d = dx * dx + dy * dy + dz * dz;
        if (d < bd0)      { bd2 = bd1; bi2 = bi1; bd1 = bd0; bi1 = bi0; bd0 = d; bi0 = j; }
        else if (d < bd1) { bd2 = bd1; bi2 = bi1; bd1 = d;   bi1 = j; }
        else if (d < bd2) { bd2 = d;   bi2 = j; }
    }
    float w0 = 1.0f / fmaxf(bd0, 1e-10f);
    float w1 = 1.0f / fmaxf(bd1, 1e-10f);
    float w2 = 1.0f / fmaxf(bd2, 1e-10f);
    float ws = w0 + w1 + w2;
    w0 /= ws; w1 /= ws; w2 /= ws;
    const float4* F0 = (const float4*)(p2 + ((size_t)b * N2 + bi0) * C2);
    const float4* F1 = (const float4*)(p2 + ((size_t)b * N2 + bi1) * C2);
    const float4* F2 = (const float4*)(p2 + ((size_t)b * N2 + bi2) * C2);
    float4* O = (float4*)(out + ((size_t)b * N1 + n) * (size_t)(C2 + C1));
    int c4 = C2 >> 2;
    for (int c = 0; c < c4; ++c) {
        float4 u0 = F0[c], u1 = F1[c], u2 = F2[c], o4;
        o4.x = w0 * u0.x + w1 * u1.x + w2 * u2.x;
        o4.y = w0 * u0.y + w1 * u1.y + w2 * u2.y;
        o4.z = w0 * u0.z + w1 * u1.z + w2 * u2.z;
        o4.w = w0 * u0.w + w1 * u1.w + w2 * u2.w;
        O[c] = o4;
    }
    if (p1) {
        const float4* Sk = (const float4*)(p1 + ((size_t)b * N1 + n) * C1);
        int k4 = C1 >> 2;
        for (int c = 0; c < k4; ++c) O[c4 + c] = Sk[c];
    }
}

// ---------------- d1: [8, 524288] @ [524288, 128] — streaming, W read once -----
__global__ void __launch_bounds__(128)
dense1_partial(const float* __restrict__ x, const float* __restrict__ W,
               float* __restrict__ partial, int total, int chunk)
{
    const int j  = threadIdx.x;                 // 0..127 output column
    const int i0 = blockIdx.x * chunk;
    float s[BATCH];
#pragma unroll
    for (int b = 0; b < BATCH; ++b) s[b] = 0.0f;
    int iend = min(i0 + chunk, total);
#pragma unroll 4
    for (int i = i0; i < iend; ++i) {
        if ((i & 63) == 0)
            __builtin_prefetch(&W[(size_t)(i + 64) * 128 + j], 0, 1);
        float w = W[(size_t)i * 128 + j];
#pragma unroll
        for (int b = 0; b < BATCH; ++b) s[b] += x[(size_t)b * total + i] * w;
    }
    float* o = partial + (size_t)blockIdx.x * (BATCH * 128);
#pragma unroll
    for (int b = 0; b < BATCH; ++b) o[b * 128 + j] = s[b];
}

__global__ void dense1_final(const float* __restrict__ partial, float* __restrict__ net1,
                             int nblocks)
{
    int g = blockIdx.x * blockDim.x + threadIdx.x;
    if (g >= BATCH * 128) return;
    float s = 0.0f;
    for (int k = 0; k < nblocks; ++k) s += partial[(size_t)k * (BATCH * 128) + g];
    net1[g] = s;
}

// ---------------- Head: lrelu(d1) -> d2 -> softmax -----------------------------
__global__ void head_kernel(const float* __restrict__ net1, const float* __restrict__ b1,
                            const float* __restrict__ W2, const float* __restrict__ b2,
                            float* __restrict__ out)
{
    __shared__ float h[BATCH * 128];
    __shared__ float lg[BATCH * 40];
    int tid = threadIdx.x;
    for (int i = tid; i < BATCH * 128; i += 256) {
        float v = net1[i] + b1[i & 127];
        h[i] = (v >= 0.0f) ? v : ALPHA * v;
    }
    __syncthreads();
    for (int i = tid; i < BATCH * 40; i += 256) {
        int b = i / 40, o = i % 40;
        float s = b2[o];
        for (int j = 0; j < 128; ++j) s += h[b * 128 + j] * W2[j * 40 + o];
        lg[i] = s;
    }
    __syncthreads();
    if (tid < BATCH) {
        int b = tid;
        float mx = -1e30f;
        for (int o = 0; o < 40; ++o) mx = fmaxf(mx, lg[b * 40 + o]);
        float sum = 0.0f;
        for (int o = 0; o < 40; ++o) { float e = expf(lg[b * 40 + o] - mx); lg[b * 40 + o] = e; sum += e; }
        for (int o = 0; o < 40; ++o) out[b * 40 + o] = lg[b * 40 + o] / sum;
    }
}

// ================================ launch ======================================
extern "C" void kernel_launch(void* const* d_in, const int* in_sizes, int n_in,
                              void* d_out, int out_size, void* d_ws, size_t ws_size,
                              hipStream_t stream)
{
    (void)in_sizes; (void)n_in; (void)out_size; (void)ws_size;

    // ---- unpack inputs (setup_inputs dict order) ----
    int p = 0;
    const float* xyz = (const float*)d_in[p++];
    const float *saw[4][3], *sab[4][3];
    for (int i = 0; i < 4; ++i)
        for (int l = 0; l < 3; ++l) { saw[i][l] = (const float*)d_in[p++]; sab[i][l] = (const float*)d_in[p++]; }
    const float *fpw[4][2], *fpb[4][2];
    for (int i = 0; i < 4; ++i)
        for (int l = 0; l < 2; ++l) { fpw[i][l] = (const float*)d_in[p++]; fpb[i][l] = (const float*)d_in[p++]; }
    const float* d1w = (const float*)d_in[p++];
    const float* d1b = (const float*)d_in[p++];
    const float* d2w = (const float*)d_in[p++];
    const float* d2b = (const float*)d_in[p++];

    // ---- workspace layout ----
    float* ws = (float*)d_ws;
    size_t off = 0;
    auto alloc = [&](size_t n) { float* q = ws + off; off += (n + 63) & ~63ull; return q; };

    float* lx[4];  lx[0] = alloc(BATCH * 1024 * 3); lx[1] = alloc(BATCH * 256 * 3);
                   lx[2] = alloc(BATCH * 64 * 3);   lx[3] = alloc(BATCH * 16 * 3);
    float* lp[4];  lp[0] = alloc(BATCH * 1024 * 64);  lp[1] = alloc(BATCH * 256 * 128);
                   lp[2] = alloc(BATCH * 64 * 256);   lp[3] = alloc(BATCH * 16 * 512);
    float* f3 = alloc((size_t)BATCH * 64 * 256);
    float* f2 = alloc((size_t)BATCH * 256 * 256);
    float* f1 = alloc((size_t)BATCH * 1024 * 256);
    float* f0 = alloc((size_t)BATCH * 2048 * 256);
    float* net1    = alloc(BATCH * 128);
    float* partial = alloc((size_t)128 * BATCH * 128);
    int*   fidx = (int*)alloc(BATCH * 1024);
    int*   bidx = (int*)alloc((size_t)BATCH * 1024 * 32);
    float* SA = alloc((size_t)BATCH * 1024 * 32 * 64);   // 16.78M floats ping
    float* SB = alloc((size_t)BATCH * 1024 * 32 * 64);   // pong

    static const int   SA_NP[4]     = {1024, 256, 64, 16};
    static const float SA_R2[4]     = {0.01f, 0.04f, 0.16f, 0.64f};
    static const int   SA_MLP[4][3] = {{32,32,64},{64,64,128},{128,128,256},{256,256,512}};
    static const int   SA_CPAD[4]   = {32, 96, 160, 288};   // (3+Cprev) padded to 32

    // ---- set-abstraction stages ----
    const float* cur_xyz = xyz;  int curN = NPTS;
    const float* cur_fe  = nullptr; int curC = 0;
    for (int i = 0; i < 4; ++i) {
        int S = SA_NP[i];
        fps_kernel<<<BATCH, 256, 0, stream>>>(cur_xyz, curN, S, fidx, lx[i]);
        ballq_kernel<<<cdiv(BATCH * S, 256), 256, 0, stream>>>(lx[i], cur_xyz, S, curN, SA_R2[i], bidx);
        group_kernel<<<cdiv(BATCH * S * 32, 256), 256, 0, stream>>>(
            cur_xyz, cur_fe, lx[i], bidx, S, curN, curC, SA_CPAD[i], SA);

        int M = BATCH * S * 32;
        float* a = SA; float* b = SB;
        int Ct = 3 + curC, Cp = SA_CPAD[i];
        for (int l = 0; l < 3; ++l) {
            int d = SA_MLP[i][l];
            dim3 grid(M / 64, d / 32);
            gemm_bias_lrelu<<<grid, 128, 0, stream>>>(a, saw[i][l], sab[i][l], b, Ct, Cp, d);
            float* t = a; a = b; b = t;
            Ct = d; Cp = d;
        }
        maxpool_kernel<<<cdiv(BATCH * S * (Ct / 4), 256), 256, 0, stream>>>(a, S, 32, Ct, lp[i]);
        cur_xyz = lx[i]; curN = S; cur_fe = lp[i]; curC = Ct;
    }

    // ---- feature-propagation stages ----
    struct FP { const float *x1, *x2, *p1, *p2; int N1, N2, C1, C2; float* out; };
    FP fps_[4] = {
        { lx[2], lx[3], lp[2], lp[3], 64,   16,   256, 512, f3 },
        { lx[1], lx[2], lp[1], f3,    256,  64,   128, 256, f2 },
        { lx[0], lx[1], lp[0], f2,    1024, 256,  64,  256, f1 },
        { xyz,   lx[0], nullptr, f1,  2048, 1024, 0,   256, f0 },
    };
    for (int i = 0; i < 4; ++i) {
        FP& F = fps_[i];
        fp_interp_kernel<<<cdiv(BATCH * F.N1, 256), 256, 0, stream>>>(
            F.x1, F.x2, F.p2, F.p1, F.N1, F.N2, F.C2, F.C1, SA);
        int M = BATCH * F.N1, c = F.C1 + F.C2;      // 768/384/320/256, all %32==0
        dim3 g0(M / 64, 256 / 32);
        gemm_bias_lrelu<<<g0, 128, 0, stream>>>(SA, fpw[i][0], fpb[i][0], SB, c, c, 256);
        gemm_bias_lrelu<<<g0, 128, 0, stream>>>(SB, fpw[i][1], fpb[i][1], F.out, 256, 256, 256);
    }

    // ---- classifier head ----
    const int total = NPTS * 256;           // 524288
    const int nblk  = 128, chunk = total / nblk;
    dense1_partial<<<nblk, 128, 0, stream>>>(f0, d1w, partial, total, chunk);
    dense1_final<<<cdiv(BATCH * 128, 256), 256, 0, stream>>>(partial, net1, nblk);
    head_kernel<<<1, 256, 0, stream>>>(net1, d1b, d2w, d2b, (float*)d_out);
}
